// ContrastiveLoss_180388627323
// MI455X (gfx1250) — compile-verified, matrix operand order checked by reading, not run on hardware
//
#include <hip/hip_runtime.h>

typedef __attribute__((ext_vector_type(16))) __bf16 v16bf;
typedef __attribute__((ext_vector_type(8)))  __bf16 v8bf;
typedef __attribute__((ext_vector_type(8)))  float  v8f;
typedef __attribute__((ext_vector_type(4)))  int    v4i;

static constexpr int   N_ROWS = 4096;
static constexpr int   DIM    = 1024;
static constexpr int   TWO_N  = 8192;
static constexpr float INV_T  = 2.0f;   // 1 / TEMPERATURE, T = 0.5

#if defined(__AMDGCN__) && __has_builtin(__builtin_amdgcn_global_load_async_to_lds_b128)
#define NT_HAS_ASYNC 1
#else
#define NT_HAS_ASYNC 0
#endif

#if defined(__AMDGCN__) && __has_builtin(__builtin_amdgcn_s_wait_asynccnt)
#define NT_ASYNC_WAIT() __builtin_amdgcn_s_wait_asynccnt(0)
#elif defined(__AMDGCN__) && NT_HAS_ASYNC
#define NT_ASYNC_WAIT() asm volatile("s_wait_asynccnt 0x0" ::: "memory")
#else
#define NT_ASYNC_WAIT() ((void)0)
#endif

// ---------------------------------------------------------------------------
// Kernel 1: row L2-normalize -> split-bf16 planes (hi + lo); zero denom[row].
// ---------------------------------------------------------------------------
__global__ __launch_bounds__(256)
void nt_normalize(const float* __restrict__ emb_i,
                  const float* __restrict__ emb_j,
                  __bf16* __restrict__ zhi,
                  __bf16* __restrict__ zlo,
                  float*  __restrict__ denom)
{
    const int row = blockIdx.x;
    const int t   = threadIdx.x;
    const float* src = (row < N_ROWS) ? (emb_i + (size_t)row * DIM)
                                      : (emb_j + (size_t)(row - N_ROWS) * DIM);
    float x[4];
    float ss = 0.f;
#pragma unroll
    for (int i = 0; i < 4; ++i) {
        x[i] = src[t + i * 256];
        ss = fmaf(x[i], x[i], ss);
    }
    __shared__ float red[256];
    red[t] = ss;
    __syncthreads();
    for (int s = 128; s > 0; s >>= 1) {
        if (t < s) red[t] += red[t + s];
        __syncthreads();
    }
    const float inv = 1.0f / fmaxf(sqrtf(red[0]), 1e-12f);
#pragma unroll
    for (int i = 0; i < 4; ++i) {
        const float  z = x[i] * inv;
        const __bf16 h = (__bf16)z;
        const float  l = z - (float)h;
        const size_t o = (size_t)row * DIM + t + i * 256;
        zhi[o] = h;
        zlo[o] = (__bf16)l;
    }
    if (t == 0) denom[row] = 0.0f;
}

// ---------------------------------------------------------------------------
// Async (or sync fallback) staging of one 128x32 bf16 K-tile pair (A rows,
// B cols) x {hi, lo} planes into one LDS buffer. 256 threads, 8 x b128 each.
// ---------------------------------------------------------------------------
__device__ __forceinline__
void nt_stage(const __bf16* __restrict__ zhi, const __bf16* __restrict__ zlo,
              int rowBase, int colBase, int k0, int t,
              __bf16 (*As)[128][40], __bf16 (*Bs)[128][40])
{
#pragma unroll
    for (int p = 0; p < 2; ++p) {
        const __bf16* gsrc = p ? zlo : zhi;
#pragma unroll
        for (int rep = 0; rep < 2; ++rep) {
            const int idx = rep * 256 + t;
            const int r   = idx >> 2;
            const int seg = idx & 3;
            const __bf16* ga = gsrc + (size_t)(rowBase + r) * DIM + k0 + seg * 8;
            const __bf16* gb = gsrc + (size_t)(colBase + r) * DIM + k0 + seg * 8;
            __bf16* la = &As[p][r][seg * 8];
            __bf16* lb = &Bs[p][r][seg * 8];
#if NT_HAS_ASYNC
            __builtin_amdgcn_global_load_async_to_lds_b128(
                (__attribute__((address_space(1))) v4i*)ga,
                (__attribute__((address_space(3))) v4i*)la, 0, 0);
            __builtin_amdgcn_global_load_async_to_lds_b128(
                (__attribute__((address_space(1))) v4i*)gb,
                (__attribute__((address_space(3))) v4i*)lb, 0, 0);
#else
            *(uint4*)la = *(const uint4*)ga;
            *(uint4*)lb = *(const uint4*)gb;
#endif
        }
    }
}

// ---------------------------------------------------------------------------
// Kernel 2: fused sim GEMM + exp-rowsum, double-buffered async LDS staging.
// Grid (64, 64): block (x=colBlk, y=rowBlk) -> 128x128 tile of sim = Z*Z^T
// (split-bf16: hi*hi + hi*lo + lo*hi), epilogue accumulates
// sum_{j != i} exp(sim/T) into denom[row] via f32 atomics.
// ---------------------------------------------------------------------------
__global__ __launch_bounds__(256)
void nt_sim_exp(const __bf16* __restrict__ zhi,
                const __bf16* __restrict__ zlo,
                float* __restrict__ denom)
{
    // [buf][plane][row][k], rows padded to 40 bf16 (80 B); 80 KB total.
    __shared__ __align__(16) __bf16 As[2][2][128][40];
    __shared__ __align__(16) __bf16 Bs[2][2][128][40];

    const int t    = threadIdx.x;
    const int lane = t & 31;
    const int wave = t >> 5;
    const int wrow = wave & 3;   // 4 row-groups of 32 rows
    const int wcol = wave >> 2;  // 2 col-groups of 64 cols
    const int ml   = lane & 15;
    const int half = lane >> 4;

    const int rowBase = blockIdx.y * 128;
    const int colBase = blockIdx.x * 128;

    v8f acc[2][4];
#pragma unroll
    for (int mt = 0; mt < 2; ++mt)
#pragma unroll
        for (int nt = 0; nt < 4; ++nt)
            acc[mt][nt] = (v8f){0.f, 0.f, 0.f, 0.f, 0.f, 0.f, 0.f, 0.f};

    // Prologue: stage K-tile 0 into buffer 0.
    nt_stage(zhi, zlo, rowBase, colBase, 0, t, As[0], Bs[0]);
    NT_ASYNC_WAIT();
    __syncthreads();

    for (int k0 = 0; k0 < DIM; k0 += 32) {
        const int cur = (k0 >> 5) & 1;
        // Prefetch next K-tile into the other buffer (hidden under WMMAs).
        if (k0 + 32 < DIM)
            nt_stage(zhi, zlo, rowBase, colBase, k0 + 32, t, As[cur ^ 1], Bs[cur ^ 1]);

#pragma unroll
        for (int mt = 0; mt < 2; ++mt) {
            const int arow = wrow * 32 + mt * 16 + ml;
            // 16-bit A operand layout (ISA 7.12.2): lane half 0 -> K {0..7,16..23},
            // lane half 1 -> K {8..15,24..31}; M = lane % 16.
            const v8bf a0h = *(const v8bf*)&As[cur][0][arow][half * 8];
            const v8bf a1h = *(const v8bf*)&As[cur][0][arow][16 + half * 8];
            const v8bf a0l = *(const v8bf*)&As[cur][1][arow][half * 8];
            const v8bf a1l = *(const v8bf*)&As[cur][1][arow][16 + half * 8];
            const v16bf ahi = __builtin_shufflevector(a0h, a1h, 0,1,2,3,4,5,6,7,8,9,10,11,12,13,14,15);
            const v16bf alo = __builtin_shufflevector(a0l, a1l, 0,1,2,3,4,5,6,7,8,9,10,11,12,13,14,15);
#pragma unroll
            for (int nt = 0; nt < 4; ++nt) {
                const int bcol = wcol * 64 + nt * 16 + ml;
                const v8bf b0h = *(const v8bf*)&Bs[cur][0][bcol][half * 8];
                const v8bf b1h = *(const v8bf*)&Bs[cur][0][bcol][16 + half * 8];
                const v8bf b0l = *(const v8bf*)&Bs[cur][1][bcol][half * 8];
                const v8bf b1l = *(const v8bf*)&Bs[cur][1][bcol][16 + half * 8];
                const v16bf bhi = __builtin_shufflevector(b0h, b1h, 0,1,2,3,4,5,6,7,8,9,10,11,12,13,14,15);
                const v16bf blo = __builtin_shufflevector(b0l, b1l, 0,1,2,3,4,5,6,7,8,9,10,11,12,13,14,15);

                v8f c = acc[mt][nt];
                c = __builtin_amdgcn_wmma_f32_16x16x32_bf16(false, ahi, false, bhi, (short)0, c, false, false);
                c = __builtin_amdgcn_wmma_f32_16x16x32_bf16(false, ahi, false, blo, (short)0, c, false, false);
                c = __builtin_amdgcn_wmma_f32_16x16x32_bf16(false, alo, false, bhi, (short)0, c, false, false);
                acc[mt][nt] = c;
            }
        }

        // Prefetch complete + all waves done reading `cur` before next iter.
        NT_ASYNC_WAIT();
        __syncthreads();
    }

    // Epilogue: exp(sim/T), mask exact diagonal, rowsum over this tile's 64
    // columns per row, atomically accumulate into denom[row].
    // C/D layout: VGPR v, lanes 0-15 -> M = v (N = lane), lanes 16-31 -> M = 8+v.
#pragma unroll
    for (int mt = 0; mt < 2; ++mt) {
#pragma unroll
        for (int v = 0; v < 8; ++v) {
            const int gr = rowBase + wrow * 32 + mt * 16 + half * 8 + v;
            float rs = 0.f;
#pragma unroll
            for (int nt = 0; nt < 4; ++nt) {
                const int gc = colBase + wcol * 64 + nt * 16 + ml;
                const float e = __expf(acc[mt][nt][v] * INV_T);
                rs += (gr == gc) ? 0.f : e;
            }
            rs += __shfl_xor(rs, 1, 32);
            rs += __shfl_xor(rs, 2, 32);
            rs += __shfl_xor(rs, 4, 32);
            rs += __shfl_xor(rs, 8, 32);
            if (ml == 0) atomicAdd(&denom[gr], rs);
        }
    }
}

// ---------------------------------------------------------------------------
// Kernel 3: positives pos[r] = dot(z_i[r], z_j[r]) in f32 (hi+lo reconstruct).
// ---------------------------------------------------------------------------
__global__ __launch_bounds__(256)
void nt_pos(const __bf16* __restrict__ zhi,
            const __bf16* __restrict__ zlo,
            float* __restrict__ pos)
{
    const int r = blockIdx.x;
    const int t = threadIdx.x;
    const __bf16* hi_i = zhi + (size_t)r * DIM;
    const __bf16* lo_i = zlo + (size_t)r * DIM;
    const __bf16* hi_j = zhi + (size_t)(r + N_ROWS) * DIM;
    const __bf16* lo_j = zlo + (size_t)(r + N_ROWS) * DIM;
    float s = 0.f;
    for (int d = t; d < DIM; d += 256) {
        const float zi = (float)hi_i[d] + (float)lo_i[d];
        const float zj = (float)hi_j[d] + (float)lo_j[d];
        s = fmaf(zi, zj, s);
    }
    __shared__ float red[256];
    red[t] = s;
    __syncthreads();
    for (int w = 128; w > 0; w >>= 1) {
        if (t < w) red[t] += red[t + w];
        __syncthreads();
    }
    if (t == 0) pos[r] = red[0];
}

// ---------------------------------------------------------------------------
// Kernel 4: loss = ( sum_r log(denom[r]) - (2/T) * sum_r pos[r] ) / (2N)
// ---------------------------------------------------------------------------
__global__ __launch_bounds__(256)
void nt_final(const float* __restrict__ denom,
              const float* __restrict__ pos,
              float* __restrict__ out)
{
    const int t = threadIdx.x;
    float s = 0.f;
    for (int r = t; r < TWO_N; r += 256) s += __logf(denom[r]);
    for (int r = t; r < N_ROWS; r += 256) s -= (2.0f * INV_T) * pos[r];
    __shared__ float red[256];
    red[t] = s;
    __syncthreads();
    for (int w = 128; w > 0; w >>= 1) {
        if (t < w) red[t] += red[t + w];
        __syncthreads();
    }
    if (t == 0) out[0] = red[0] / (float)TWO_N;
}

// ---------------------------------------------------------------------------
extern "C" void kernel_launch(void* const* d_in, const int* in_sizes, int n_in,
                              void* d_out, int out_size, void* d_ws, size_t ws_size,
                              hipStream_t stream)
{
    (void)in_sizes; (void)n_in; (void)out_size; (void)ws_size;
    const float* emb_i = (const float*)d_in[0];
    const float* emb_j = (const float*)d_in[1];
    float* out = (float*)d_out;

    // Workspace: zhi[8192][1024] bf16 | zlo[8192][1024] bf16 |
    //            denom[8192] f32 | pos[4096] f32   (~33.6 MB)
    __bf16* zhi  = (__bf16*)d_ws;
    __bf16* zlo  = zhi + (size_t)TWO_N * DIM;
    float* denom = (float*)(zlo + (size_t)TWO_N * DIM);
    float* pos   = denom + TWO_N;

    nt_normalize<<<TWO_N, 256, 0, stream>>>(emb_i, emb_j, zhi, zlo, denom);
    nt_sim_exp<<<dim3(64, 64), 256, 0, stream>>>(zhi, zlo, denom);
    nt_pos<<<N_ROWS, 256, 0, stream>>>(zhi, zlo, pos);
    nt_final<<<1, 256, 0, stream>>>(denom, pos, out);
}
